// ResidualAttentionBlock_66108136620269
// MI455X (gfx1250) — matrix operand order repaired and checked by reading, not compiled, not run on hardware
//
#include <hip/hip_runtime.h>

typedef __attribute__((ext_vector_type(16))) _Float16 v16h;
typedef __attribute__((ext_vector_type(8)))  float    v8f;
typedef __attribute__((ext_vector_type(4)))  int      v4i;

#define WMMA_F16(A, B, C) \
  __builtin_amdgcn_wmma_f32_16x16x32_f16(false, (A), false, (B), (short)0, (C), false, false)

__device__ inline v8f vzero8() { v8f z; for (int i = 0; i < 8; i++) z[i] = 0.f; return z; }

// ---------------- async global->LDS copy (CDNA5), with sync fallback ----------
#if defined(__has_builtin)
#if __has_builtin(__builtin_amdgcn_global_load_async_to_lds_b128)
#define HAVE_ASYNC_LDS 1
#endif
#endif

typedef __attribute__((address_space(1))) v4i gv4i;  // global 16B chunk
typedef __attribute__((address_space(3))) v4i lv4i;  // LDS 16B chunk

__device__ inline void cp16(const _Float16* g, _Float16* l) {
#ifdef HAVE_ASYNC_LDS
  __builtin_amdgcn_global_load_async_to_lds_b128((gv4i*)g, (lv4i*)l, 0, 0);
#else
  *(uint4*)l = *(const uint4*)g;
#endif
}

__device__ inline void wait_async0() {
#if defined(__has_builtin)
#if __has_builtin(__builtin_amdgcn_s_wait_asynccnt)
  __builtin_amdgcn_s_wait_asynccnt(0);
#else
  asm volatile("s_wait_asynccnt 0x0" ::: "memory");
#endif
#else
  asm volatile("s_wait_asynccnt 0x0" ::: "memory");
#endif
}

// ---------------------------------------------------------------- constants
constexpr int BB = 128, NN = 197, CC = 768, HH = 12, HD = 64;
constexpr int LK = 138, NP = 139;          // kept tokens, kept+CLS
constexpr int M1 = BB * NN;                // 25216 rows (full tokens)
constexpr int M2 = BB * NP;                // 17792 rows (pruned tokens)

// ---------------------------------------------------------------- misc kernels
__global__ __launch_bounds__(256) void f2h_kernel(const float* __restrict__ in,
                                                  _Float16* __restrict__ out, int n) {
  int i = blockIdx.x * 256 + threadIdx.x;
  if (i < n) out[i] = (_Float16)in[i];
}

__global__ __launch_bounds__(256) void zero_kernel(float* p, int n) {
  int i = blockIdx.x * 256 + threadIdx.x;
  if (i < n) p[i] = 0.f;
}

// LayerNorm over C=768; one block per row; writes f16
__global__ __launch_bounds__(256) void ln_kernel(const float* __restrict__ x,
                                                 const float* __restrict__ g,
                                                 const float* __restrict__ be,
                                                 _Float16* __restrict__ out) {
  __shared__ float red[256];
  int row = blockIdx.x, t = threadIdx.x;
  const float* xr = x + (size_t)row * CC;
  float vals[3], s = 0.f;
  for (int i = 0; i < 3; i++) { vals[i] = xr[t + i * 256]; s += vals[i]; }
  red[t] = s; __syncthreads();
  for (int o = 128; o > 0; o >>= 1) { if (t < o) red[t] += red[t + o]; __syncthreads(); }
  float mean = red[0] / (float)CC; __syncthreads();
  float s2 = 0.f;
  for (int i = 0; i < 3; i++) { float d = vals[i] - mean; s2 += d * d; }
  red[t] = s2; __syncthreads();
  for (int o = 128; o > 0; o >>= 1) { if (t < o) red[t] += red[t + o]; __syncthreads(); }
  float rinv = rsqrtf(red[0] / (float)CC + 1e-5f);
  for (int i = 0; i < 3; i++) {
    int c = t + i * 256;
    out[(size_t)row * CC + c] = (_Float16)((vals[i] - mean) * rinv * g[c] + be[c]);
  }
}

// ---------------------------------------------------------------- WMMA GEMM
// C[M,N] = A[M,K](f16) @ B[K,N](f16), fp32 accumulate. Double-buffered LDS,
// async global->LDS staging overlapped with WMMA compute.
// MODE 0: split into q(*hd^-1/2)/k/v f16 [B,H,N,64]
// MODE 1: outf = resid + acc + bias        (proj)
// MODE 2: outh = quickGELU(acc + bias) f16 (fc)
// MODE 3: outf += acc + bias               (cproj into d_out)
template <int MODE>
__global__ __launch_bounds__(256)
void gemm_kernel(const _Float16* __restrict__ A, const _Float16* __restrict__ Bw,
                 int M, int N, int K,
                 const float* __restrict__ bias, const float* __restrict__ resid,
                 float* __restrict__ outf, _Float16* __restrict__ outh,
                 _Float16* __restrict__ qh, _Float16* __restrict__ kh,
                 _Float16* __restrict__ vh) {
  __shared__ _Float16 As[2][128][40];   // +8 pad, double buffered
  __shared__ _Float16 Bs[2][32][136];   // +8 pad, double buffered
  const int t = threadIdx.x, lane = t & 31, wave = t >> 5;
  const int wr = wave >> 2, wc = wave & 3;
  const int row0 = blockIdx.y * 128, col0 = blockIdx.x * 128;
  const int rA = t >> 1, sA = (t & 1) * 16;
  const int rB = t >> 3, sB = (t & 7) * 16;

  v8f acc[4][2];
  for (int i = 0; i < 4; i++) for (int j = 0; j < 2; j++) acc[i][j] = vzero8();

  const _Float16* gA = A + (size_t)(row0 + rA) * K + sA;
  const _Float16* gB = Bw + (size_t)rB * N + col0 + sB;

  const int nk = K >> 5;
  // stage tile 0
  cp16(gA, &As[0][rA][sA]);
  cp16(gA + 8, &As[0][rA][sA + 8]);
  cp16(gB, &Bs[0][rB][sB]);
  cp16(gB + 8, &Bs[0][rB][sB + 8]);

  for (int kk = 0; kk < nk; kk++) {
    wait_async0();
    __syncthreads();                    // tile kk resident in buf kk&1
    if (kk + 1 < nk) {                  // issue loads for next tile
      int k0 = (kk + 1) << 5, bu = (kk + 1) & 1;
      cp16(gA + k0, &As[bu][rA][sA]);
      cp16(gA + k0 + 8, &As[bu][rA][sA + 8]);
      cp16(gB + (size_t)k0 * N, &Bs[bu][rB][sB]);
      cp16(gB + (size_t)k0 * N + 8, &Bs[bu][rB][sB + 8]);
    }
    const int cu = kk & 1;
    v16h bfrag[2];
    for (int jn = 0; jn < 2; jn++) {    // B: lane%16 = N col, K in same order as A
      int cb = wc * 32 + jn * 16;
      for (int s = 0; s < 16; s++)
        bfrag[jn][s] = Bs[cu][(lane >> 4) * 8 + (s < 8 ? s : s + 8)][cb + (lane & 15)];
    }
    for (int im = 0; im < 4; im++) {
      int ar = wr * 64 + im * 16 + (lane & 15);
      int ko = (lane >> 4) * 8;
      v16h afrag;  // A: K pairs 0..7 then 16..23 (lane<16) / 8..15,24..31 (lane>=16)
      ((uint4*)&afrag)[0] = *(const uint4*)&As[cu][ar][ko];
      ((uint4*)&afrag)[1] = *(const uint4*)&As[cu][ar][16 + ko];
      for (int jn = 0; jn < 2; jn++) acc[im][jn] = WMMA_F16(afrag, bfrag[jn], acc[im][jn]);
    }
    // no trailing barrier: buf kk&1 is only overwritten by loads issued after
    // the NEXT top barrier, which every wave reaches only after its WMMAs.
  }

  const int ncol = lane & 15, mb = (lane >> 4) * 8;
  for (int im = 0; im < 4; im++)
    for (int jn = 0; jn < 2; jn++)
      for (int g = 0; g < 8; g++) {
        int m = row0 + wr * 64 + im * 16 + mb + g;
        int n = col0 + wc * 32 + jn * 16 + ncol;
        float v = acc[im][jn][g];
        if (MODE == 0) {
          int b = m / NN, tok = m % NN;
          int which = n / CC, rem = n % CC;
          int hh = rem >> 6, d = rem & 63;
          size_t idx = (((size_t)b * HH + hh) * NN + tok) * HD + d;
          if (which == 0)      qh[idx] = (_Float16)(v * 0.125f);  // hd^-0.5
          else if (which == 1) kh[idx] = (_Float16)v;
          else                 vh[idx] = (_Float16)v;
        } else if (MODE == 1) {
          outf[(size_t)m * N + n] = resid[(size_t)m * N + n] + v + bias[n];
        } else if (MODE == 2) {
          float u = v + bias[n];
          outh[(size_t)m * N + n] = (_Float16)(u / (1.f + __expf(-1.702f * u)));
        } else {
          outf[(size_t)m * N + n] += v + bias[n];
        }
      }
}

// ---------------------------------------------------------------- attention
// grid.x = B*H, grid.y = 7 (32 q rows each). Dynamic LDS, KT/V aliased.
constexpr int ATT_SMEM = 4608 /*Qs 32x72*/ + 32256 /*max(KT 64x232, Vs 224x72)*/ +
                         28672 /*S 32x224 f32*/ + 14336 /*Ph 32x224*/ +
                         1024 /*red*/ + 128 /*rstat*/;

__global__ __launch_bounds__(256)
void attn_kernel(const _Float16* __restrict__ qh, const _Float16* __restrict__ kh,
                 const _Float16* __restrict__ vh, const int* __restrict__ mask,
                 _Float16* __restrict__ outh, float* __restrict__ cls) {
  extern __shared__ char smem[];
  _Float16* Qs = (_Float16*)smem;                          // [32][72]
  _Float16* KT = (_Float16*)(smem + 4608);                 // [64][232]
  _Float16* Vs = KT;                                       // [224][72] (aliased)
  float*    S  = (float*)(smem + 4608 + 32256);            // [32][224]
  _Float16* Ph = (_Float16*)(smem + 4608 + 32256 + 28672); // [32][224]
  float*   red = (float*)(smem + 4608 + 32256 + 28672 + 14336); // [32][8]
  float* rstat = red + 256;                                // [32]

  const int t = threadIdx.x, lane = t & 31, wave = t >> 5;
  const int bh = blockIdx.x, b = bh / HH, h = bh % HH;
  const int qt = blockIdx.y;

  // stage Q via async copy (zero-pad rows >= 197)
  {
    const _Float16* qg = qh + (size_t)bh * NN * HD;
    int e = t * 8, r = e >> 6, c = e & 63, gr = qt * 32 + r;
    if (gr < NN) cp16(qg + (size_t)gr * HD + c, &Qs[r * 72 + c]);
    else { uint4 z; z.x = z.y = z.z = z.w = 0u; *(uint4*)&Qs[r * 72 + c] = z; }
  }
  // stage K transposed (VALU path), pad keys 197..223
  {
    const _Float16* kg = kh + (size_t)bh * NN * HD;
    for (int e = t; e < NN * HD; e += 256) { int r = e >> 6, d = e & 63; KT[d * 232 + r] = kg[e]; }
    for (int e = t; e < 27 * HD; e += 256) { int key = NN + (e >> 6), d = e & 63; KT[d * 232 + key] = (_Float16)0.f; }
  }
  wait_async0();
  __syncthreads();

  // scores: 2 m-tiles x 13 key-tiles (keys 0..207)
  for (int t2 = wave; t2 < 26; t2 += 8) {
    int im = t2 / 13, ik = t2 % 13;
    v8f acc = vzero8();
    for (int dk = 0; dk < 2; dk++) {
      v16h a, bf;
      int ar = im * 16 + (lane & 15);
      int ko = dk * 32 + ((lane >> 4) * 8);
      ((uint4*)&a)[0] = *(const uint4*)&Qs[ar * 72 + ko];
      ((uint4*)&a)[1] = *(const uint4*)&Qs[ar * 72 + ko + 16];
      int br = dk * 32 + ((lane >> 4) * 8);
      for (int s = 0; s < 16; s++)
        bf[s] = KT[(br + (s < 8 ? s : s + 8)) * 232 + ik * 16 + (lane & 15)];
      acc = WMMA_F16(a, bf, acc);
    }
    int ncol = lane & 15, mb = (lane >> 4) * 8;
    for (int g = 0; g < 8; g++) S[(im * 16 + mb + g) * 224 + ik * 16 + ncol] = acc[g];
  }
  __syncthreads();

  // masked softmax, 8 threads per row
  {
    int row = t >> 3, sub = t & 7;
    const int* mrow = mask + (size_t)b * NN * NN;  // row 0 of [N,N]
    float mx = -3.4e38f;
    for (int c = sub; c < NN; c += 8) {
      float v = (mrow[c] != 0) ? S[row * 224 + c] : -1e9f;
      S[row * 224 + c] = v;
      mx = fmaxf(mx, v);
    }
    red[row * 8 + sub] = mx; __syncthreads();
    if (sub == 0) { float m = red[row * 8]; for (int i = 1; i < 8; i++) m = fmaxf(m, red[row * 8 + i]); rstat[row] = m; }
    __syncthreads();
    float rmax = rstat[row], sum = 0.f;
    for (int c = sub; c < NN; c += 8) { float e = __expf(S[row * 224 + c] - rmax); S[row * 224 + c] = e; sum += e; }
    red[row * 8 + sub] = sum; __syncthreads();
    if (sub == 0) { float s = 0.f; for (int i = 0; i < 8; i++) s += red[row * 8 + i]; rstat[row] = s; }
    __syncthreads();
    float rinv = 1.f / rstat[row];
    bool isrow0 = (qt == 0 && row == 0);
    for (int c = sub; c < 224; c += 8) {
      float p = (c < NN) ? S[row * 224 + c] * rinv : 0.f;
      Ph[row * 224 + c] = (_Float16)p;
      if (isrow0 && c >= 1 && c < NN) atomicAdd(&cls[(size_t)b * (NN - 1) + (c - 1)], p * (1.0f / (float)HH));
    }
  }
  __syncthreads();

  // stage V via async copy (aliases KT region; issued only after all KT reads)
  {
    const _Float16* vg = vh + (size_t)bh * NN * HD;
    for (int ch = t; ch < (NN * HD) / 8; ch += 256) {  // 1576 chunks of 8 halfs
      int e = ch * 8, r = e >> 6, d = e & 63;
      cp16(vg + e, &Vs[r * 72 + d]);
    }
    for (int e = t; e < 27 * HD; e += 256) { int r = NN + (e >> 6), d = e & 63; Vs[r * 72 + d] = (_Float16)0.f; }
  }
  wait_async0();
  __syncthreads();

  // out = P @ V : 2 m-tiles x 4 d-tiles, one per wave
  {
    int im = wave >> 2, id = wave & 3;
    v8f acc = vzero8();
    for (int kb = 0; kb < 7; kb++) {
      v16h a, bf;
      int ar = im * 16 + (lane & 15);
      int ko = kb * 32 + ((lane >> 4) * 8);
      ((uint4*)&a)[0] = *(const uint4*)&Ph[ar * 224 + ko];
      ((uint4*)&a)[1] = *(const uint4*)&Ph[ar * 224 + ko + 16];
      int br = kb * 32 + ((lane >> 4) * 8);
      for (int s = 0; s < 16; s++)
        bf[s] = Vs[(br + (s < 8 ? s : s + 8)) * 72 + id * 16 + (lane & 15)];
      acc = WMMA_F16(a, bf, acc);
    }
    int ncol = lane & 15, mb = (lane >> 4) * 8;
    for (int g = 0; g < 8; g++) {
      int qrow = qt * 32 + im * 16 + mb + g;
      if (qrow < NN)
        outh[((size_t)b * NN + qrow) * CC + h * HD + id * 16 + ncol] = (_Float16)acc[g];
    }
  }
}

// ---------------------------------------------------------------- top-k + gather
__global__ __launch_bounds__(256)
void topk_gather_kernel(const float* __restrict__ cls, const int* __restrict__ mask,
                        const float* __restrict__ x2, float* __restrict__ outx,
                        int* __restrict__ attn_row) {
  __shared__ float sc[NN - 1];
  __shared__ unsigned char taken[NN - 1];
  __shared__ int idxs[LK];
  __shared__ float redv[256];
  __shared__ int redi[256];
  __shared__ int s_cnt;
  const int b = blockIdx.x, t = threadIdx.x;
  if (t == 0) s_cnt = 0;
  __syncthreads();
  if (t < NN - 1) {
    bool valid = mask[(size_t)b * NN * NN + (1 + t)] != 0;
    sc[t] = valid ? cls[(size_t)b * (NN - 1) + t] : -3.4e38f;
    taken[t] = 0;
    if (valid) atomicAdd(&s_cnt, 1);
  }
  __syncthreads();
  const int left = (int)ceilf((float)s_cnt * 0.7f);
  for (int it = 0; it < LK; it++) {  // argmax, tie -> lowest index (matches top_k)
    float v; int idx;
    if (t < NN - 1 && !taken[t]) { v = sc[t]; idx = t; }
    else { v = -3.4e38f; idx = 1 << 29; }
    redv[t] = v; redi[t] = idx; __syncthreads();
    for (int o = 128; o > 0; o >>= 1) {
      if (t < o) {
        float v2 = redv[t + o]; int i2 = redi[t + o];
        if (v2 > redv[t] || (v2 == redv[t] && i2 < redi[t])) { redv[t] = v2; redi[t] = i2; }
      }
      __syncthreads();
    }
    if (t == 0) { idxs[it] = redi[0]; if (redi[0] < NN - 1) taken[redi[0]] = 1; }
    __syncthreads();
  }
  float* xo = outx + (size_t)b * NP * CC;
  const float* xb = x2 + (size_t)b * NN * CC;
  for (int c = t; c < CC; c += 256) xo[c] = xb[c];  // CLS row
  if (t == 0) attn_row[b * NP] = 1;
  for (int j = 0; j < LK; j++) {
    bool keep = j < left;
    int src = (keep && idxs[j] < NN - 1) ? idxs[j] : 0;
    for (int c = t; c < CC; c += 256) {
      float v = keep ? xb[(size_t)(1 + src) * CC + c] : 0.f;
      xo[(size_t)(1 + j) * CC + c] = v;
      if (c == 0) attn_row[b * NP + 1 + j] = (v == 0.f) ? 0 : 1;
    }
  }
}

__global__ __launch_bounds__(256) void bcast_mask_kernel(const int* __restrict__ attn_row,
                                                         int* __restrict__ outm) {
  size_t i = (size_t)blockIdx.x * 256 + threadIdx.x;
  constexpr size_t total = (size_t)BB * NP * NP;
  if (i < total) {
    int b = (int)(i / (NP * NP));
    int j = (int)(i % NP);
    outm[i] = attn_row[b * NP + j];
  }
}

// ---------------------------------------------------------------- workspace layout
constexpr size_t AL(size_t x) { return (x + 255) & ~(size_t)255; }
constexpr size_t SZ_WQKVH  = AL((size_t)CC * 3 * CC * 2);
constexpr size_t SZ_WPROJH = AL((size_t)CC * CC * 2);
constexpr size_t SZ_WFCH   = AL((size_t)CC * 4 * CC * 2);
constexpr size_t SZ_WCPH   = AL((size_t)4 * CC * CC * 2);
constexpr size_t SZ_H1     = AL((size_t)M1 * CC * 2);        // h1 -> attn_out -> h2
constexpr size_t SZ_QKV1   = AL((size_t)BB * HH * NN * HD * 2);
constexpr size_t SZ_X2     = AL((size_t)M1 * CC * 4);
constexpr size_t SZ_CLS    = AL((size_t)BB * (NN - 1) * 4);
constexpr size_t SZ_AROW   = AL((size_t)BB * NP * 4);

constexpr size_t OFF_WQKVH  = 0;
constexpr size_t OFF_WPROJH = OFF_WQKVH + SZ_WQKVH;
constexpr size_t OFF_WFCH   = OFF_WPROJH + SZ_WPROJH;
constexpr size_t OFF_WCPH   = OFF_WFCH + SZ_WFCH;
constexpr size_t OFF_H1     = OFF_WCPH + SZ_WCPH;
constexpr size_t OFF_Q      = OFF_H1 + SZ_H1;
constexpr size_t OFF_K      = OFF_Q + SZ_QKV1;
constexpr size_t OFF_V      = OFF_K + SZ_QKV1;
constexpr size_t OFF_X2     = OFF_V + SZ_QKV1;
constexpr size_t OFF_CLS    = OFF_X2 + SZ_X2;
constexpr size_t OFF_AROW   = OFF_CLS + SZ_CLS;
// reuse: u_h (17792*3072 f16 = 109.3MB) lives in the q/k/v region (116.2MB)
constexpr size_t OFF_UH     = OFF_Q;

extern "C" void kernel_launch(void* const* d_in, const int* in_sizes, int n_in,
                              void* d_out, int out_size, void* d_ws, size_t ws_size,
                              hipStream_t stream) {
  const float* x       = (const float*)d_in[0];
  const int*   amask   = (const int*)d_in[1];
  const float* ln1_g   = (const float*)d_in[2];
  const float* ln1_b   = (const float*)d_in[3];
  const float* w_qkv   = (const float*)d_in[4];
  const float* w_proj  = (const float*)d_in[5];
  const float* b_proj  = (const float*)d_in[6];
  const float* ln2_g   = (const float*)d_in[7];
  const float* ln2_b   = (const float*)d_in[8];
  const float* w_fc    = (const float*)d_in[9];
  const float* b_fc    = (const float*)d_in[10];
  const float* w_cproj = (const float*)d_in[11];
  const float* b_cproj = (const float*)d_in[12];

  char* ws = (char*)d_ws;
  _Float16* w_qkv_h  = (_Float16*)(ws + OFF_WQKVH);
  _Float16* w_proj_h = (_Float16*)(ws + OFF_WPROJH);
  _Float16* w_fc_h   = (_Float16*)(ws + OFF_WFCH);
  _Float16* w_cp_h   = (_Float16*)(ws + OFF_WCPH);
  _Float16* h1_h     = (_Float16*)(ws + OFF_H1);   // also attn_out_h, h2_h
  _Float16* q_h      = (_Float16*)(ws + OFF_Q);
  _Float16* k_h      = (_Float16*)(ws + OFF_K);
  _Float16* v_h      = (_Float16*)(ws + OFF_V);
  float*    x2       = (float*)(ws + OFF_X2);
  float*    cls      = (float*)(ws + OFF_CLS);
  int*      arow     = (int*)(ws + OFF_AROW);
  _Float16* u_h      = (_Float16*)(ws + OFF_UH);

  float* out_x = (float*)d_out;                       // [B,139,768]
  int*   out_m = (int*)((float*)d_out + (size_t)BB * NP * CC);  // [B,139,139]

  // weights fp32 -> f16
  auto cvt = [&](const float* src, _Float16* dst, int n) {
    f2h_kernel<<<(n + 255) / 256, 256, 0, stream>>>(src, dst, n);
  };
  cvt(w_qkv, w_qkv_h, CC * 3 * CC);
  cvt(w_proj, w_proj_h, CC * CC);
  cvt(w_fc, w_fc_h, CC * 4 * CC);
  cvt(w_cproj, w_cp_h, 4 * CC * CC);
  zero_kernel<<<(BB * (NN - 1) + 255) / 256, 256, 0, stream>>>(cls, BB * (NN - 1));

  // ln1 -> h1 (f16)
  ln_kernel<<<M1, 256, 0, stream>>>(x, ln1_g, ln1_b, h1_h);

  // qkv = h1 @ w_qkv -> q(*0.125)/k/v  [B,H,N,64] f16
  gemm_kernel<0><<<dim3(3 * CC / 128, M1 / 128), 256, 0, stream>>>(
      h1_h, w_qkv_h, M1, 3 * CC, CC, nullptr, nullptr, nullptr, nullptr, q_h, k_h, v_h);

  // attention (writes attn_out into h1 region, cls scores via atomics)
  attn_kernel<<<dim3(BB * HH, 7), 256, ATT_SMEM, stream>>>(q_h, k_h, v_h, amask, h1_h, cls);

  // x2 = x + attn_out @ w_proj + b_proj
  gemm_kernel<1><<<dim3(CC / 128, M1 / 128), 256, 0, stream>>>(
      h1_h, w_proj_h, M1, CC, CC, b_proj, x, x2, nullptr, nullptr, nullptr, nullptr);

  // top-k prune + gather -> out_x, attn row flags
  topk_gather_kernel<<<BB, 256, 0, stream>>>(cls, amask, x2, out_x, arow);
  bcast_mask_kernel<<<(int)(((size_t)BB * NP * NP + 255) / 256), 256, 0, stream>>>(arow, out_m);

  // ln2 on pruned x -> h2 (f16, reuses h1 region)
  ln_kernel<<<M2, 256, 0, stream>>>(out_x, ln2_g, ln2_b, h1_h);

  // u = quickGELU(h2 @ w_fc + b_fc) f16 (reuses q/k/v region)
  gemm_kernel<2><<<dim3(4 * CC / 128, M2 / 128), 256, 0, stream>>>(
      h1_h, w_fc_h, M2, 4 * CC, CC, b_fc, nullptr, nullptr, u_h, nullptr, nullptr, nullptr);

  // out_x += u @ w_cproj + b_cproj
  gemm_kernel<3><<<dim3(CC / 128, M2 / 128), 256, 0, stream>>>(
      u_h, w_cp_h, M2, CC, 4 * CC, b_cproj, nullptr, out_x, nullptr, nullptr, nullptr, nullptr);

  (void)in_sizes; (void)n_in; (void)out_size; (void)ws_size;
}
